// VQVAE_11879879544402
// MI455X (gfx1250) — matrix operand order repaired and checked by reading, not compiled, not run on hardware
//
#include <hip/hip_runtime.h>

typedef _Float16 half_t;
typedef __attribute__((ext_vector_type(16))) _Float16 v16h;
typedef __attribute__((ext_vector_type(8)))  _Float16 v8h;
typedef __attribute__((ext_vector_type(4)))  _Float16 v4h;
typedef __attribute__((ext_vector_type(8)))  float    v8f;
typedef __attribute__((ext_vector_type(4)))  float    v4f;
typedef __attribute__((ext_vector_type(4)))  unsigned int v4u;
typedef __attribute__((ext_vector_type(4)))  int      v4i;
typedef __attribute__((ext_vector_type(8)))  int      v8i;

#define N_ROWS 32768
#define DIMS   256
#define KCODES 2048
#define QSIZE  (N_ROWS * DIMS)

#define BROWS    64        // rows of x per workgroup
#define XSTRIDE  264       // LDS x row stride in halves (padded, 16B-divisible)
#define TROWS    64        // codebook rows per TDM tile
#define BSTRIDE  264       // LDS B row stride in halves: 512B data + 16B TDM pad
#define NTILES   (KCODES / TROWS)   // 32

#define WMMA_F16(A, B, C) \
    __builtin_amdgcn_wmma_f32_16x16x32_f16(false, (A), false, (B), (short)0, (C), false, false)

// --------- TDM: async copy one 64x256 f16 codebook tile global->LDS --------
// D# per CDNA5 ISA 8.3/8.4: 2D tensor, data_size=1 (2B),
// pad_interval code 6 (=128 DWORDs = 512B), pad_amount code 3 (=4 DWORDs = 16B)
// -> each 512B row followed by 16B pad => LDS row stride 528B (= BSTRIDE halves).
__device__ __forceinline__ void tdm_load_tile(unsigned lds_byte_off,
                                              const half_t* gsrc) {
    unsigned long long ga = (unsigned long long)(uintptr_t)gsrc;
    v4u g0;
    g0[0] = 1u;                                   // count=1 (valid user descriptor)
    g0[1] = lds_byte_off;                         // LDS dest (bytes)
    g0[2] = (unsigned)(ga & 0xFFFFFFFFu);         // global_addr[31:0]
    g0[3] = (unsigned)((ga >> 32) & 0x01FFFFFFu)  // global_addr[56:32]
          | (2u << 30);                           // type=2 ("image")
    v8i g1;
    g1[0] = (1 << 16)        // data_size = 1 -> 2 bytes
          | (1 << 20)        // pad_enable
          | (6 << 22)        // pad_interval: 128 DWORDs
          | (3 << 25);       // pad_amount:   4 DWORDs
    g1[1] = (int)((DIMS & 0xFFFF) << 16);                    // tensor_dim0[15:0]
    g1[2] = (int)(((DIMS >> 16) & 0xFFFF) | ((KCODES & 0xFFFF) << 16)); // dim0 hi | dim1 lo
    g1[3] = (int)(((KCODES >> 16) & 0xFFFF) | ((DIMS & 0xFFFF) << 16)); // dim1 hi | tile_dim0
    g1[4] = TROWS;                                           // tile_dim1 | tile_dim2=0
    g1[5] = DIMS;                                            // tensor_dim0_stride[31:0]
    g1[6] = 0;
    g1[7] = 0;
    v4i z4 = {0, 0, 0, 0};
#if __clang_major__ >= 23
    v8i z8 = {0, 0, 0, 0, 0, 0, 0, 0};
    __builtin_amdgcn_tensor_load_to_lds(g0, g1, z4, z4, z8, 0);
#else
    __builtin_amdgcn_tensor_load_to_lds(g0, g1, z4, z4, 0);
#endif
}

// ---------------- prep: codebook f32 -> f16, c_sq, zero SSE accumulator ----
__global__ void vq_prep_kernel(const float* __restrict__ cb,
                               float* __restrict__ wsf,      // [0]=sse accum, [4..4+K)=c_sq
                               half_t* __restrict__ cbh) {
    int row = blockIdx.x;           // 0..2047
    int tid = threadIdx.x;          // 0..63
    const float* src = cb + (size_t)row * DIMS;
    half_t*      dst = cbh + (size_t)row * DIMS;

    int c = tid * 4;
    v4f v = *(const v4f*)(src + c);
    v4h h = { (half_t)v[0], (half_t)v[1], (half_t)v[2], (half_t)v[3] };
    *(v4h*)(dst + c) = h;
    float s = v[0]*v[0] + v[1]*v[1] + v[2]*v[2] + v[3]*v[3];

    for (int m = 16; m; m >>= 1) s += __shfl_xor(s, m, 32);
    __shared__ float wsum[2];
    if ((tid & 31) == 0) wsum[tid >> 5] = s;
    __syncthreads();
    if (tid == 0) {
        wsf[4 + row] = wsum[0] + wsum[1];
        if (row == 0) wsf[0] = 0.0f;   // zero SSE accumulator each launch
    }
}

// ---------------- main: TDM-fed WMMA distances + argmin + gather + SSE -----
__global__ __launch_bounds__(128)
void vq_main_kernel(const float*  __restrict__ x,
                    const float*  __restrict__ cb,
                    const half_t* __restrict__ cbh,
                    const float*  __restrict__ csq,
                    float*        __restrict__ sse_accum,
                    float*        __restrict__ out_q,
                    float*        __restrict__ out_idx) {
    __shared__ __align__(128) half_t xs[BROWS * XSTRIDE];
    __shared__ __align__(128) half_t btile[2 * TROWS * BSTRIDE];  // double-buffered B
    __shared__ float  cs_s[KCODES];
    __shared__ int    idx_s[BROWS];
    __shared__ float  wsum[4];

    const int tid     = threadIdx.x;
    const int rowBase = blockIdx.x * BROWS;
    const int wave    = tid >> 5;
    const int lane    = tid & 31;
    const int m       = lane & 15;   // A row within tile / B column within tile
    const int hsel    = lane >> 4;   // lane-half select
    const int waveRow = wave * 16;

    // Kick off TDM for tile 0 into buffer 0 (wave 0 only; EXEC-independent).
    if (wave == 0) {
        tdm_load_tile((unsigned)(uintptr_t)&btile[0], cbh);
    }

    // Phase 1a: stage 64 rows of x into LDS as f16 (8 elems / iteration)
    for (int chunk = tid; chunk < (BROWS * DIMS) / 8; chunk += 128) {
        int row = chunk >> 5;                 // 32 chunks of 8 per 256-wide row
        int c8  = (chunk & 31) * 8;
        const float* src = x + (size_t)(rowBase + row) * DIMS + c8;
        v4f a = *(const v4f*)(src);
        v4f b = *(const v4f*)(src + 4);
        v8h h = { (half_t)a[0], (half_t)a[1], (half_t)a[2], (half_t)a[3],
                  (half_t)b[0], (half_t)b[1], (half_t)b[2], (half_t)b[3] };
        *(v8h*)(&xs[row * XSTRIDE + c8]) = h;
    }
    // Phase 1b: stage c_sq into LDS (2048 floats)
    for (int i = tid * 4; i < KCODES; i += 128 * 4) {
        *(v4f*)&cs_s[i] = *(const v4f*)&csq[i];
    }
    __syncthreads();

    // Preload all 8 A fragments (covers D=256, K=32 each).
    // A layout (16-bit 16x32): element e -> K = hsel*16 + e
    v16h afrag[8];
    const half_t* xrow = &xs[(waveRow + m) * XSTRIDE];
#pragma unroll
    for (int kb = 0; kb < 8; ++kb) {
        v8h lo = *(const v8h*)(xrow + kb * 32 + hsel * 16);
        v8h hi = *(const v8h*)(xrow + kb * 32 + hsel * 16 + 8);
        afrag[kb] = __builtin_shufflevector(lo, hi, 0,1,2,3,4,5,6,7,8,9,10,11,12,13,14,15);
    }

    float best[8];
    int   bidx[8];
#pragma unroll
    for (int r = 0; r < 8; ++r) { best[r] = 3.4e38f; bidx[r] = 0; }

    // Sweep codebook in 32 TDM-prefetched tiles of 64 entries.
    // ||x||^2 is argmin-invariant; dist = c_sq - 2*x.c
    for (int t = 0; t < NTILES; ++t) {
        if (wave == 0) {
            if (t + 1 < NTILES) {
                tdm_load_tile((unsigned)(uintptr_t)&btile[((t + 1) & 1) * (TROWS * BSTRIDE)],
                              cbh + (size_t)(t + 1) * TROWS * DIMS);
                __builtin_amdgcn_s_wait_tensorcnt(1);   // tile t complete
            } else {
                __builtin_amdgcn_s_wait_tensorcnt(0);
            }
        }
        __syncthreads();   // tile t visible to all waves

        const int tileBase = t * TROWS;
        // B layout (16-bit 32x16): lane column = m ; element e -> K = hsel*16 + e
        const half_t* q0 = &btile[(t & 1) * (TROWS * BSTRIDE)] + (size_t)m * BSTRIDE + hsel * 16;
        const half_t* q1 = q0 + (size_t)16 * BSTRIDE;
        const half_t* q2 = q0 + (size_t)32 * BSTRIDE;
        const half_t* q3 = q0 + (size_t)48 * BSTRIDE;

        v8f acc0 = {}, acc1 = {}, acc2 = {}, acc3 = {};
#pragma unroll
        for (int kb = 0; kb < 8; ++kb) {
            v16h b0 = *(const v16h*)(q0 + kb * 32);
            v16h b1 = *(const v16h*)(q1 + kb * 32);
            v16h b2 = *(const v16h*)(q2 + kb * 32);
            v16h b3 = *(const v16h*)(q3 + kb * 32);
            acc0 = WMMA_F16(afrag[kb], b0, acc0);
            acc1 = WMMA_F16(afrag[kb], b1, acc1);
            acc2 = WMMA_F16(afrag[kb], b2, acc2);
            acc3 = WMMA_F16(afrag[kb], b3, acc3);
        }

        float cs0 = cs_s[tileBase + m];
        float cs1 = cs_s[tileBase + 16 + m];
        float cs2 = cs_s[tileBase + 32 + m];
        float cs3 = cs_s[tileBase + 48 + m];
#pragma unroll
        for (int r = 0; r < 8; ++r) {
            float d0 = fmaf(-2.0f, acc0[r], cs0);
            if (d0 < best[r]) { best[r] = d0; bidx[r] = tileBase + m; }
            float d1 = fmaf(-2.0f, acc1[r], cs1);
            if (d1 < best[r]) { best[r] = d1; bidx[r] = tileBase + 16 + m; }
            float d2 = fmaf(-2.0f, acc2[r], cs2);
            if (d2 < best[r]) { best[r] = d2; bidx[r] = tileBase + 32 + m; }
            float d3 = fmaf(-2.0f, acc3[r], cs3);
            if (d3 < best[r]) { best[r] = d3; bidx[r] = tileBase + 48 + m; }
        }
        __syncthreads();   // tile t fully consumed before its buffer is rewritten
    }

    // Cross-lane argmin over the 16 columns; masks 1..8 stay within each half.
#pragma unroll
    for (int mask = 8; mask >= 1; mask >>= 1) {
#pragma unroll
        for (int r = 0; r < 8; ++r) {
            float od = __shfl_xor(best[r], mask, 32);
            int   oi = __shfl_xor(bidx[r], mask, 32);
            if (od < best[r] || (od == best[r] && oi < bidx[r])) {
                best[r] = od; bidx[r] = oi;
            }
        }
    }
    if (m == 0) {
#pragma unroll
        for (int r = 0; r < 8; ++r) {
            int rl = waveRow + hsel * 8 + r;   // C VGPR r: rows r / r+8 per half
            idx_s[rl] = bidx[r];
            out_idx[rowBase + rl] = (float)bidx[r];
        }
    }
    __syncthreads();

    // Phase 4: f32 gather, straight-through output x + (q - x), SSE partials
    float sse = 0.0f;
    for (int c = tid * 4; c < BROWS * DIMS; c += 128 * 4) {
        int row = c >> 8;
        int col = c & 255;
        int kk  = idx_s[row];
        v4f q  = *(const v4f*)(cb + (size_t)kk * DIMS + col);
        v4f xv = *(const v4f*)(x + (size_t)(rowBase + row) * DIMS + col);
        v4f st;
#pragma unroll
        for (int j = 0; j < 4; ++j) {
            float dq = q[j] - xv[j];
            st[j] = xv[j] + dq;
            sse = fmaf(dq, dq, sse);
        }
        *(v4f*)(out_q + (size_t)(rowBase + row) * DIMS + col) = st;
    }
    for (int mask = 16; mask; mask >>= 1) sse += __shfl_xor(sse, mask, 32);
    if (lane == 0) wsum[wave] = sse;
    __syncthreads();
    if (tid == 0) atomicAdd(sse_accum, wsum[0] + wsum[1] + wsum[2] + wsum[3]);
}

// ---------------- finalize: loss = 1.25 * SSE / (N*D) ----------------------
__global__ void vq_finalize_kernel(const float* __restrict__ sse_accum,
                                   float* __restrict__ out_loss) {
    out_loss[0] = sse_accum[0] * 1.25f / (float)QSIZE;
}

extern "C" void kernel_launch(void* const* d_in, const int* in_sizes, int n_in,
                              void* d_out, int out_size, void* d_ws, size_t ws_size,
                              hipStream_t stream) {
    (void)in_sizes; (void)n_in; (void)out_size; (void)ws_size;
    const float* x  = (const float*)d_in[0];   // [32,32,32,256] f32
    const float* cb = (const float*)d_in[1];   // [2048,256] f32

    float*  wsf = (float*)d_ws;                        // [0]=sse, [4..4+K)=c_sq
    float*  csq = wsf + 4;
    half_t* cbh = (half_t*)((char*)d_ws + 16384);      // f16 codebook, 1 MB

    float* out      = (float*)d_out;
    float* out_q    = out;                  // QSIZE floats
    float* out_loss = out + QSIZE;          // 1 float
    float* out_idx  = out + QSIZE + 1;      // N_ROWS floats (indices)

    vq_prep_kernel<<<KCODES, 64, 0, stream>>>(cb, wsf, cbh);
    vq_main_kernel<<<N_ROWS / BROWS, 128, 0, stream>>>(x, cb, cbh, csq,
                                                       wsf, out_q, out_idx);
    vq_finalize_kernel<<<1, 1, 0, stream>>>(wsf, out_loss);
}
